// ModernHopfieldMemory_49065706389516
// MI455X (gfx1250) — matrix operand order repaired and checked by reading, not compiled, not run on hardware
//
#include <hip/hip_runtime.h>

typedef __attribute__((ext_vector_type(16))) __bf16 v16bf;
typedef __attribute__((ext_vector_type(8)))  __bf16 v8bf;
typedef __attribute__((ext_vector_type(8)))  float  v8f;
typedef __attribute__((ext_vector_type(4)))  float  f32x4;
typedef __attribute__((ext_vector_type(4)))  unsigned int u32x4;

#define D_DIM      512
#define N_PAT      65536
#define B_QRY      256
#define BLK_N      2048
#define N_CHUNKS   (N_PAT / BLK_N)   // 32
#define M_BLK      32                // queries per block (2 WMMA M-tiles per wave)
#define Q_TILES    (B_QRY / M_BLK)   // 8
#define LDS_STRIDE 2052              // 2048 + 4 pad: half-wave row offset lands 32 banks apart
#define TOPK       32

// ---- fragment loaders ------------------------------------------------------

// Pack 16 f32 into a bf16x16 fragment; casts lower to v_cvt_pk_bf16_f32.
__device__ __forceinline__ v16bf cvt16(f32x4 f0, f32x4 f1, f32x4 f2, f32x4 f3) {
  v16bf r;
#pragma unroll
  for (int i = 0; i < 4; ++i) {
    r[i]      = (__bf16)f0[i];
    r[4 + i]  = (__bf16)f1[i];
    r[8 + i]  = (__bf16)f2[i];
    r[12 + i] = (__bf16)f3[i];
  }
  return r;
}

// f32 source: two 32B chunks -> one fragment.
__device__ __forceinline__ v16bf load_frag_f32(const float* __restrict__ p0,
                                               const float* __restrict__ p1) {
  return cvt16(*(const f32x4*)(p0), *(const f32x4*)(p0 + 4),
               *(const f32x4*)(p1), *(const f32x4*)(p1 + 4));
}

// bf16 source: two 16B chunks -> one fragment, no conversion.
union BfFrag { v16bf v; u32x4 u[2]; };
__device__ __forceinline__ v16bf load_frag_bf16(const __bf16* p0, const __bf16* p1) {
  BfFrag r;
  r.u[0] = *(const u32x4*)(p0);
  r.u[1] = *(const u32x4*)(p1);
  return r.v;
}

// ---- one-time f32 -> bf16 conversion (fast path) ---------------------------

__global__ void __launch_bounds__(256) cvt_f32_bf16(const float* __restrict__ src,
                                                    __bf16* __restrict__ dst,
                                                    long long n) {
  const long long i = ((long long)blockIdx.x * 256 + threadIdx.x) * 8;
  if (i + 8 <= n) {
    f32x4 a = *(const f32x4*)(src + i);
    f32x4 b = *(const f32x4*)(src + i + 4);
    v8bf r;
#pragma unroll
    for (int j = 0; j < 4; ++j) { r[j] = (__bf16)a[j]; r[4 + j] = (__bf16)b[j]; }
    *(v8bf*)(dst + i) = r;
  }
}

// ---- kernel 1: WMMA score tile (32 queries x 2048 patterns) + local top-32 --
// A fragment (16x32): lane half 0 holds K {c0..c0+7, c1..c1+7}; half 1 the +8 offsets.
// B fragment (32x16): lane = column N, same K order as A (half h: K 8h..8h+7, 16+8h..16+8h+7).

template <bool BF16WS>
__global__ void __launch_bounds__(512) hop_scores_topk(
    const void* __restrict__ qptr, const void* __restrict__ pptr,
    float* __restrict__ pS, int* __restrict__ pI)
{
  extern __shared__ float sc[];                 // [M_BLK][LDS_STRIDE]
  const int tid   = threadIdx.x;
  const int wave  = tid >> 5;
  const int lane  = tid & 31;
  const int half  = lane >> 4;
  const int l16   = lane & 15;
  const int chunk = blockIdx.x;                 // 0..31
  const int qtile = blockIdx.y;                 // 0..7
  const int chunkBase = chunk * BLK_N;
  const int qbase     = qtile * M_BLK;
  const int nwbase    = chunkBase + wave * 128; // this wave's 128-pattern slab

  v8f acc0[8] = {};
  v8f acc1[8] = {};
  for (int kk = 0; kk < D_DIM; kk += 32) {
    const int c0 = kk + 8 * half;
    const int c1 = kk + 16 + 8 * half;
    v16bf a0, a1;
    if constexpr (BF16WS) {
      const __bf16* q0 = (const __bf16*)qptr + (size_t)(qbase + l16) * D_DIM;
      a0 = load_frag_bf16(q0 + c0, q0 + c1);
      a1 = load_frag_bf16(q0 + (size_t)16 * D_DIM + c0, q0 + (size_t)16 * D_DIM + c1);
    } else {
      const float* q0 = (const float*)qptr + (size_t)(qbase + l16) * D_DIM;
      a0 = load_frag_f32(q0 + c0, q0 + c1);
      a1 = load_frag_f32(q0 + (size_t)16 * D_DIM + c0, q0 + (size_t)16 * D_DIM + c1);
    }
#pragma unroll
    for (int t = 0; t < 8; ++t) {
      v16bf b;
      if constexpr (BF16WS) {
        const __bf16* pb = (const __bf16*)pptr +
            (size_t)(nwbase + t * 16 + l16) * D_DIM + kk + 8 * half;
        b = load_frag_bf16(pb, pb + 16);
      } else {
        const float* pb = (const float*)pptr +
            (size_t)(nwbase + t * 16 + l16) * D_DIM + kk + 8 * half;
        b = load_frag_f32(pb, pb + 16);
      }
      acc0[t] = __builtin_amdgcn_wmma_f32_16x16x32_bf16(
          false, a0, false, b, (short)0, acc0[t], false, false);
      acc1[t] = __builtin_amdgcn_wmma_f32_16x16x32_bf16(
          false, a1, false, b, (short)0, acc1[t], false, false);
    }
  }

  // C layout: lane half 0 -> rows j, half 1 -> rows j+8; N = lane&15.
#pragma unroll
  for (int t = 0; t < 8; ++t) {
    const int col = wave * 128 + t * 16 + l16;
#pragma unroll
    for (int j = 0; j < 8; ++j) {
      sc[(j + 8 * half) * LDS_STRIDE + col]      = acc0[t][j];
      sc[(16 + j + 8 * half) * LDS_STRIDE + col] = acc1[t][j];
    }
  }
  __syncthreads();

  // Each wave selects top-32 for rows {wave, wave+16} over the 2048-wide chunk.
#pragma unroll 1
  for (int rr = 0; rr < 2; ++rr) {
    const int r = wave + 16 * rr;
    volatile float* row = sc + r * LDS_STRIDE;
    const size_t obase = (((size_t)qtile * N_CHUNKS + chunk) * M_BLK + r) * TOPK;
    for (int it = 0; it < TOPK; ++it) {
      float bs = -__builtin_inff();
      int   bi = 0x7fffffff;
      for (int c = lane; c < BLK_N; c += 32) {
        float v = row[c];
        if (v > bs || (v == bs && c < bi)) { bs = v; bi = c; }
      }
#pragma unroll
      for (int off = 16; off > 0; off >>= 1) {
        float os = __shfl_xor(bs, off, 32);
        int   oi = __shfl_xor(bi, off, 32);
        if (os > bs || (os == bs && oi < bi)) { bs = os; bi = oi; }
      }
      if (lane == 0) {
        row[bi] = -__builtin_inff();
        pS[obase + it] = bs;
        pI[obase + it] = chunkBase + bi;
      }
    }
  }
}

// ---- kernel 2: merge partial top-32 -> exact top-32, softmax, gather -------

__global__ void __launch_bounds__(128) hop_merge_gather(
    const float* __restrict__ patterns, const float* __restrict__ pS,
    const int* __restrict__ pI, float* __restrict__ out)
{
  __shared__ float s[1024];
  __shared__ int   id[1024];
  __shared__ float wsel[TOPK];
  __shared__ int   isel[TOPK];

  const int b     = blockIdx.x;
  const int qtile = b / M_BLK;
  const int row   = b % M_BLK;
  const int tid   = threadIdx.x;

  for (int i = tid; i < 1024; i += 128) {
    const int c = i >> 5, k = i & 31;
    const size_t o = (((size_t)qtile * N_CHUNKS + c) * M_BLK + row) * TOPK + k;
    s[i]  = pS[o];
    id[i] = pI[o];
  }
  __syncthreads();

  if (tid < 32) {
    const int lane = tid;
    volatile float* vs = s;
    for (int it = 0; it < TOPK; ++it) {
      float bs = -__builtin_inff();
      int   bi = 0, bid = 0x7fffffff;
      for (int c = lane; c < 1024; c += 32) {
        float v  = vs[c];
        int   gi = id[c];
        if (v > bs || (v == bs && gi < bid)) { bs = v; bi = c; bid = gi; }
      }
#pragma unroll
      for (int off = 16; off > 0; off >>= 1) {
        float os = __shfl_xor(bs, off, 32);
        int   oi = __shfl_xor(bi, off, 32);
        int   od = __shfl_xor(bid, off, 32);
        if (os > bs || (os == bs && od < bid)) { bs = os; bi = oi; bid = od; }
      }
      if (lane == 0) { vs[bi] = -__builtin_inff(); wsel[it] = bs; isel[it] = bid; }
    }
    // softmax over the 32 selected scores (wsel[0] is the max, extracted first)
    float m = wsel[0];
    float e = __expf(wsel[lane] - m);
    float sum = e;
#pragma unroll
    for (int off = 16; off > 0; off >>= 1) sum += __shfl_xor(sum, off, 32);
    wsel[lane] = e / sum;
  }
  __syncthreads();

  for (int d = tid; d < D_DIM; d += 128) {
    float a = 0.f;
#pragma unroll 8
    for (int k = 0; k < TOPK; ++k)
      a += wsel[k] * patterns[(size_t)isel[k] * D_DIM + d];
    out[(size_t)b * D_DIM + d] = a;
  }
}

// ---- launcher --------------------------------------------------------------

extern "C" void kernel_launch(void* const* d_in, const int* in_sizes, int n_in,
                              void* d_out, int out_size, void* d_ws, size_t ws_size,
                              hipStream_t stream) {
  const float* query    = (const float*)d_in[0];
  const float* patterns = (const float*)d_in[1];
  float* out = (float*)d_out;

  const size_t nPartial = (size_t)Q_TILES * N_CHUNKS * M_BLK * TOPK;   // 262144
  const size_t partialBytes = nPartial * (sizeof(float) + sizeof(int)); // 2 MB
  const size_t patElems = (size_t)N_PAT * D_DIM;                        // 33.5 M
  const size_t qryElems = (size_t)B_QRY * D_DIM;                        // 131072
  const size_t bfBytes  = (patElems + qryElems) * 2;                    // ~64.3 MB

  dim3 g1(N_CHUNKS, Q_TILES);
  const size_t lds1 = (size_t)M_BLK * LDS_STRIDE * sizeof(float);       // ~257 KB of 320 KB

  if (ws_size >= bfBytes + partialBytes) {
    // Fast path: pre-convert inputs to bf16, GEMM streams half the bytes from L2.
    __bf16* bpat = (__bf16*)d_ws;
    __bf16* bqry = (__bf16*)((char*)d_ws + patElems * 2);
    float*  pS   = (float*)((char*)d_ws + bfBytes);
    int*    pI   = (int*)(pS + nPartial);

    cvt_f32_bf16<<<dim3((unsigned)(patElems / 8 / 256)), dim3(256), 0, stream>>>(
        patterns, bpat, (long long)patElems);
    cvt_f32_bf16<<<dim3((unsigned)(qryElems / 8 / 256)), dim3(256), 0, stream>>>(
        query, bqry, (long long)qryElems);
    hop_scores_topk<true><<<g1, dim3(512), lds1, stream>>>(bqry, bpat, pS, pI);
    hop_merge_gather<<<dim3(B_QRY), dim3(128), 0, stream>>>(patterns, pS, pI, out);
  } else {
    // Fallback: convert in-loop from f32 (2x L2 bytes, extra VALU).
    float* pS = (float*)d_ws;
    int*   pI = (int*)((char*)d_ws + nPartial * sizeof(float));
    hop_scores_topk<false><<<g1, dim3(512), lds1, stream>>>(query, patterns, pS, pI);
    hop_merge_gather<<<dim3(B_QRY), dim3(128), 0, stream>>>(patterns, pS, pI, out);
  }
}
